// LSTMDiscriminator_74577812128389
// MI455X (gfx1250) — compile-verified, matrix operand order checked by reading, not run on hardware
//
#include <hip/hip_runtime.h>
#include <hip/hip_bf16.h>
#include <stdint.h>

// ---------------------------------------------------------------------------
// MI455X / gfx1250 LSTM discriminator.
//   Kernel 1: x_proj GEMM [B*T,200]x[200,64] via v_wmma_f32_16x16x32_bf16,
//             split-bf16 (truncated hi + residual lo, packed with v_perm_b32)
//             for ~fp32 accuracy; memory-bound (210 MB @ 23.3 TB/s ~ 9 us).
//   Kernel 2: 16 independent single-wave workgroups scan T=1024 steps each
//             (batch-tiled recurrence). Per step: 12 gate WMMAs + 3 head
//             WMMAs; h relayout via LDS; A-fragment carried across the
//             head/next-recurrence boundary; v_tanh_f32 activations.
// ---------------------------------------------------------------------------

typedef __attribute__((ext_vector_type(16))) __bf16 v16bf;
typedef __attribute__((ext_vector_type(8)))  float  v8f;

union BFrag { unsigned u[8]; uint4 q[2]; v16bf v; };

static constexpr int Bdim = 256;
static constexpr int Tdim = 1024;
static constexpr int Idim = 200;
static constexpr int Hdim = 16;
static constexpr int G4H  = 64;    // 4*H gate width
static constexpr int KCH  = 7;     // ceil(200/32) K-chunks of 32
static constexpr int WKP  = 116;   // LDS pair-stride (dwords): 16B aligned, low conflict

// ---- split-bf16 helpers ---------------------------------------------------
// Truncating pack of two f32 into two bf16 (one v_perm_b32). Truncation (not
// RNE) is fine because the residual-lo product recovers the dropped bits.
__device__ __forceinline__ unsigned pk2_bf16(float f0, float f1) {
  return __builtin_amdgcn_perm(__float_as_uint(f1), __float_as_uint(f0),
                               0x07060302u);
}
__device__ __forceinline__ float bf16_trunc_f(float f) {
  return __uint_as_float(__float_as_uint(f) & 0xFFFF0000u);
}
__device__ __forceinline__ unsigned pk2_bf16_lo(float f0, float f1) {
  return pk2_bf16(f0 - bf16_trunc_f(f0), f1 - bf16_trunc_f(f1));
}

// ---- activations: gfx1250 v_tanh_f32 if the builtin exists ---------------
#if __has_builtin(__builtin_amdgcn_tanhf)
__device__ __forceinline__ float tanh_hw(float x) {
  return __builtin_amdgcn_tanhf(x);
}
#else
__device__ __forceinline__ float tanh_hw(float x) {
  float e = __expf(-2.0f * x);
  return (1.0f - e) / (1.0f + e);
}
#endif
__device__ __forceinline__ float sigf(float x) {   // 0.5 + 0.5*tanh(x/2)
  return fmaf(0.5f, tanh_hw(0.5f * x), 0.5f);
}

// 3-product split-bf16 WMMA: ~fp32 accuracy (hi*hi + hi*lo + lo*hi)
__device__ __forceinline__ v8f wmma3(const BFrag& aHi, const BFrag& aLo,
                                     const BFrag& bHi, const BFrag& bLo, v8f c) {
  c = __builtin_amdgcn_wmma_f32_16x16x32_bf16(false, aHi.v, false, bHi.v, (short)0, c, false, false);
  c = __builtin_amdgcn_wmma_f32_16x16x32_bf16(false, aHi.v, false, bLo.v, (short)0, c, false, false);
  c = __builtin_amdgcn_wmma_f32_16x16x32_bf16(false, aLo.v, false, bHi.v, (short)0, c, false, false);
  return c;
}

// ---------------------------------------------------------------------------
// Kernel 1: x_proj[(t*B + b)*64 + g] = x[b,t,:] . W_ih[g,:] + b_ih[g] + b_hh[g]
// 256 threads = 8 waves/WG; each wave owns one 16-row M-tile; W_ih staged
// once per WG in LDS as packed split-bf16, K padded 200 -> 224.
// ---------------------------------------------------------------------------
__global__ __launch_bounds__(256)
void xproj_wmma_kernel(const float* __restrict__ x,
                       const float* __restrict__ W_ih,
                       const float* __restrict__ b_ih,
                       const float* __restrict__ b_hh,
                       float* __restrict__ xp) {
  __shared__ __align__(16) unsigned w_hi[G4H * WKP];
  __shared__ __align__(16) unsigned w_lo[G4H * WKP];

  const int tid = threadIdx.x;
  // Stage W_ih as B-operand pairs: pair p of row n holds bf16(W[n,2p]),bf16(W[n,2p+1])
  for (int idx = tid; idx < G4H * 112; idx += 256) {
    int n = idx / 112, p = idx % 112;
    int k0 = 2 * p, k1 = 2 * p + 1;
    float f0 = (k0 < Idim) ? W_ih[n * Idim + k0] : 0.0f;
    float f1 = (k1 < Idim) ? W_ih[n * Idim + k1] : 0.0f;
    w_hi[n * WKP + p] = pk2_bf16(f0, f1);
    w_lo[n * WKP + p] = pk2_bf16_lo(f0, f1);
  }
  __syncthreads();

  const int lane = tid & 31, wave = tid >> 5;
  const int nl = lane & 15, hh = lane >> 4;           // column-in-tile, K-half
  const int mtile = blockIdx.x * 8 + wave;            // 16384 M-tiles total
  const float* arow = x + (size_t)(mtile * 16 + nl) * Idim;

  v8f acc[4] = {v8f{0,0,0,0,0,0,0,0}, v8f{0,0,0,0,0,0,0,0},
                v8f{0,0,0,0,0,0,0,0}, v8f{0,0,0,0,0,0,0,0}};

  #pragma unroll
  for (int kc = 0; kc < KCH; ++kc) {
    // A-fragment: lane holds K = kc*32 + {hh*8..hh*8+7, 16+hh*8..16+hh*8+7}
    float a[16];
    if (kc < 6) {
      const float* p = arow + kc * 32 + hh * 8;
      float4 q0 = *(const float4*)(p);
      float4 q1 = *(const float4*)(p + 4);
      float4 q2 = *(const float4*)(p + 16);
      float4 q3 = *(const float4*)(p + 20);
      a[0]=q0.x; a[1]=q0.y; a[2]=q0.z; a[3]=q0.w;
      a[4]=q1.x; a[5]=q1.y; a[6]=q1.z; a[7]=q1.w;
      a[8]=q2.x; a[9]=q2.y; a[10]=q2.z; a[11]=q2.w;
      a[12]=q3.x; a[13]=q3.y; a[14]=q3.z; a[15]=q3.w;
    } else { // tail chunk: K 192..223, valid only K<200
      #pragma unroll
      for (int j = 0; j < 8; ++j) {
        int k = 192 + hh * 8 + j;
        a[j]     = (k < Idim) ? arow[k] : 0.0f;
        a[j + 8] = 0.0f;                 // K >= 208 always out of range
      }
    }
    BFrag aHi, aLo;
    #pragma unroll
    for (int d = 0; d < 8; ++d) {
      aHi.u[d] = pk2_bf16(a[2 * d], a[2 * d + 1]);
      aLo.u[d] = pk2_bf16_lo(a[2 * d], a[2 * d + 1]);
    }
    #pragma unroll
    for (int nt = 0; nt < 4; ++nt) {
      const unsigned* bh = &w_hi[(nt * 16 + nl) * WKP + kc * 16 + hh * 4];
      const unsigned* bl = &w_lo[(nt * 16 + nl) * WKP + kc * 16 + hh * 4];
      BFrag bHi, bLo;
      bHi.q[0] = *(const uint4*)(bh);
      bHi.q[1] = *(const uint4*)(bh + 8);
      bLo.q[0] = *(const uint4*)(bl);
      bLo.q[1] = *(const uint4*)(bl + 8);
      acc[nt] = wmma3(aHi, aLo, bHi, bLo, acc[nt]);
    }
  }

  // Bias + store in scan-friendly [t][b][64] order (fits in 192MB global L2).
  #pragma unroll
  for (int nt = 0; nt < 4; ++nt) {
    float bias = b_ih[nt * 16 + nl] + b_hh[nt * 16 + nl];
    #pragma unroll
    for (int r = 0; r < 8; ++r) {
      int mrow = mtile * 16 + r + hh * 8;     // flat b*T + t
      int b = mrow >> 10, t = mrow & (Tdim - 1);
      xp[((size_t)(t * Bdim + b) << 6) + nt * 16 + nl] = acc[nt][r] + bias;
    }
  }
}

// ---------------------------------------------------------------------------
// Kernel 2: sequential scan. 16 workgroups x 1 wave; WG g owns batch rows
// [16g, 16g+16). Per step: gates = x_proj_t + h @ W_hh^T (12 WMMA, split-bf16,
// K padded 16->32), activations, c/h update, head (3 WMMA) + lane reduction.
// The A-fragment of h_t is built once (after publishing h_t to LDS) and used
// for both the step-t head and the step-(t+1) recurrence.
// ---------------------------------------------------------------------------
__global__ __launch_bounds__(32)
void lstm_scan_kernel(const float* __restrict__ xp,
                      const float* __restrict__ h0,
                      const float* __restrict__ c0,
                      const float* __restrict__ Whh,
                      const float* __restrict__ W1,
                      const float* __restrict__ b1,
                      const float* __restrict__ W2,
                      const float* __restrict__ b2,
                      float* __restrict__ out) {
  __shared__ __align__(16) float h_lds[16 * 16];
  const int lane = threadIdx.x;
  const int nl = lane & 15, hh = lane >> 4;
  const int base = blockIdx.x * 16;

  // Preload B-fragments: W_hh^T (4 N-tiles) and W1^T, split-bf16, K 16->32 pad.
  BFrag bhh_hi[4], bhh_lo[4], bw1_hi, bw1_lo;
  #pragma unroll
  for (int nt = 0; nt < 4; ++nt) {
    #pragma unroll
    for (int d = 0; d < 8; ++d) {
      int k0 = ((d & 4) ? 16 : 0) + hh * 8 + (d & 3) * 2;
      int ng = nt * 16 + nl;
      float f0 = (k0 < Hdim)     ? Whh[ng * Hdim + k0]     : 0.0f;
      float f1 = (k0 + 1 < Hdim) ? Whh[ng * Hdim + k0 + 1] : 0.0f;
      bhh_hi[nt].u[d] = pk2_bf16(f0, f1);
      bhh_lo[nt].u[d] = pk2_bf16_lo(f0, f1);
    }
  }
  #pragma unroll
  for (int d = 0; d < 8; ++d) {
    int k0 = ((d & 4) ? 16 : 0) + hh * 8 + (d & 3) * 2;
    float f0 = (k0 < Hdim)     ? W1[nl * Hdim + k0]     : 0.0f;
    float f1 = (k0 + 1 < Hdim) ? W1[nl * Hdim + k0 + 1] : 0.0f;
    bw1_hi.u[d] = pk2_bf16(f0, f1);
    bw1_lo.u[d] = pk2_bf16_lo(f0, f1);
  }
  const float b1v = b1[nl];
  const float w2v = W2[nl];
  const float b2v = b2[0];

  // Initial state: h0 tile -> LDS, c0 tile -> D-layout registers.
  for (int idx = lane; idx < 256; idx += 32)
    h_lds[idx] = h0[(base + (idx >> 4)) * Hdim + (idx & 15)];
  __syncthreads();

  v8f c, accum = v8f{0,0,0,0,0,0,0,0};
  #pragma unroll
  for (int r = 0; r < 8; ++r)
    c[r] = c0[(base + r + hh * 8) * Hdim + nl];

  // A-fragment builder: rows m = nl, K = 0..15 from LDS, upper K-half zero.
  BFrag aHi, aLo;
  auto load_afrag = [&](BFrag& Hi, BFrag& Lo) {
    const float* hp = &h_lds[nl * 16 + hh * 8];
    #pragma unroll
    for (int d = 0; d < 4; ++d) {
      Hi.u[d] = pk2_bf16(hp[2 * d], hp[2 * d + 1]);
      Lo.u[d] = pk2_bf16_lo(hp[2 * d], hp[2 * d + 1]);
      Hi.u[d + 4] = 0u;
      Lo.u[d + 4] = 0u;
    }
  };
  load_afrag(aHi, aLo);   // fragment of h_{t-1}, carried across iterations

  for (int t = 0; t < Tdim; ++t) {
    // gfx1250 global_prefetch_b8: pull next timestep's 4KB tile ahead of time.
    __builtin_prefetch(xp + ((size_t)((t + 1) * Bdim + base) << 6) + lane * 32, 0, 0);

    // Gate accumulators seeded with x_proj tile (contiguous 4KB, L2 resident).
    const float* xrow = xp + ((size_t)(t * Bdim + base + hh * 8) << 6) + nl;
    v8f g0, g1, g2, g3;
    #pragma unroll
    for (int r = 0; r < 8; ++r) {
      g0[r] = xrow[r * 64];
      g1[r] = xrow[r * 64 + 16];
      g2[r] = xrow[r * 64 + 32];
      g3[r] = xrow[r * 64 + 48];
    }

    g0 = wmma3(aHi, aLo, bhh_hi[0], bhh_lo[0], g0);   // i
    g1 = wmma3(aHi, aLo, bhh_hi[1], bhh_lo[1], g1);   // f
    g2 = wmma3(aHi, aLo, bhh_hi[2], bhh_lo[2], g2);   // g
    g3 = wmma3(aHi, aLo, bhh_hi[3], bhh_lo[3], g3);   // o

    v8f hD;
    #pragma unroll
    for (int r = 0; r < 8; ++r) {
      float iv = sigf(g0[r]);
      float fv = sigf(g1[r]);
      float gv = tanh_hw(g2[r]);
      float ov = sigf(g3[r]);
      float cv = fv * c[r] + iv * gv;
      c[r]  = cv;
      hD[r] = ov * tanh_hw(cv);
    }

    // Publish h_t (D-layout -> LDS); same-wave DS ops are in-order.
    #pragma unroll
    for (int r = 0; r < 8; ++r)
      h_lds[(r + hh * 8) * 16 + nl] = hD[r];

    // Rebuild the carried A-fragment once: used by the head below AND by the
    // next iteration's recurrence.
    load_afrag(aHi, aLo);

    // Head on h_t: relu(h W1^T + b1) . W2 + b2 -> sigmoid -> accumulate.
    v8f d1;
    #pragma unroll
    for (int r = 0; r < 8; ++r) d1[r] = b1v;
    d1 = wmma3(aHi, aLo, bw1_hi, bw1_lo, d1);

    float s[8];
    #pragma unroll
    for (int r = 0; r < 8; ++r) s[r] = fmaxf(d1[r], 0.0f) * w2v;
    #pragma unroll
    for (int off = 1; off < 16; off <<= 1) {
      #pragma unroll
      for (int r = 0; r < 8; ++r) s[r] += __shfl_xor(s[r], off, 32);
    }
    #pragma unroll
    for (int r = 0; r < 8; ++r) accum[r] += sigf(s[r] + b2v);
  }

  if (nl == 0) {  // one lane per K-half writes its 8 batch rows
    #pragma unroll
    for (int r = 0; r < 8; ++r)
      out[base + hh * 8 + r] = accum[r] * (1.0f / (float)Tdim);
  }
}

// ---------------------------------------------------------------------------
extern "C" void kernel_launch(void* const* d_in, const int* in_sizes, int n_in,
                              void* d_out, int out_size, void* d_ws, size_t ws_size,
                              hipStream_t stream) {
  (void)in_sizes; (void)n_in; (void)out_size; (void)ws_size;
  const float* x    = (const float*)d_in[0];
  const float* h0   = (const float*)d_in[1];
  const float* c0   = (const float*)d_in[2];
  const float* W_ih = (const float*)d_in[3];
  const float* W_hh = (const float*)d_in[4];
  const float* b_ih = (const float*)d_in[5];
  const float* b_hh = (const float*)d_in[6];
  const float* W1   = (const float*)d_in[7];
  const float* b1   = (const float*)d_in[8];
  const float* W2   = (const float*)d_in[9];
  const float* b2   = (const float*)d_in[10];
  float* xp  = (float*)d_ws;            // [T][B][64] fp32 = 67 MB scratch
  float* out = (float*)d_out;           // [B,1] fp32

  const int mtiles = (Bdim * Tdim) / 16;          // 16384
  xproj_wmma_kernel<<<mtiles / 8, 256, 0, stream>>>(x, W_ih, b_ih, b_hh, xp);
  lstm_scan_kernel<<<Bdim / 16, 32, 0, stream>>>(xp, h0, c0, W_hh, W1, b1, W2, b2, out);
}
